// HolidayComponentLightweight_25220047962526
// MI455X (gfx1250) — matrix operand order repaired and checked by reading, not compiled, not run on hardware
//
#include <hip/hip_runtime.h>
#include <hip/hip_bf16.h>

typedef _Float16 half_t;
typedef __attribute__((ext_vector_type(16))) _Float16 v16h;
typedef __attribute__((ext_vector_type(8)))  float    v8f;

#define H_    16
#define C_    32
#define HID_  512
#define SKU_  64
#define ROWS_ 32
#define INV_TEMP 1.4285714285714286f   /* 1/0.7 */
#define LN_EPS   1e-3f
#define TIME_MAX 365.0f

// ---------------------------------------------------------------------------
// helpers
// ---------------------------------------------------------------------------
__device__ __forceinline__ float fast_rcp(float x) { return __builtin_amdgcn_rcpf(x); }

// mish(x) = x * tanh(softplus(x)).  Since e^{softplus(x)} = 1 + e^x exactly,
// tanh(sp) = (t^2-1)/(t^2+1) with t = 1+e^x.  One v_exp + one v_rcp, branchless.
__device__ __forceinline__ float mish_f(float x) {
    float t  = 1.0f + __expf(x);
    float t2 = t * t;
    float y  = x * (t2 - 1.0f) * fast_rcp(t2 + 1.0f);
    return (x > 15.0f) ? x : y;     // v_cndmask; avoids inf/inf
}

__device__ __forceinline__ float sigmoid_f(float x) {
    return fast_rcp(1.0f + __expf(-x));
}

// A-fragment (f16 16x32, ISA 7.12.2): lane half selects K-block, element e ->
// vg=e>>1 ; k = (vg<4?0:16) + half*8 + (vg&3)*2 + (e&1) ; row m = lane&15
__device__ __forceinline__ v16h load_a_frag_lds(const half_t* base, int row_stride,
                                                int kc, int lane) {
    const int halfq = lane >> 4;
    const int m = lane & 15;
    v16h a;
#pragma unroll
    for (int e = 0; e < 16; ++e) {
        int vg = e >> 1;
        int k = ((vg < 4) ? 0 : 16) + halfq * 8 + ((vg & 3) << 1) + (e & 1);
        a[e] = base[m * row_stride + kc * 32 + k];
    }
    return a;
}

// Same A-fragment gather but from a global f32 matrix (for sku_embedding)
__device__ __forceinline__ v16h load_a_frag_g32(const float* base, int ld,
                                                int kc, int lane) {
    const int halfq = lane >> 4;
    const int m = lane & 15;
    v16h a;
#pragma unroll
    for (int e = 0; e < 16; ++e) {
        int vg = e >> 1;
        int k = ((vg < 4) ? 0 : 16) + halfq * 8 + ((vg & 3) << 1) + (e & 1);
        a[e] = (half_t)base[m * ld + kc * 32 + k];
    }
    return a;
}

// B-fragment from pre-packed weights: tile stores 32 lanes x 16 halfs
// contiguously -> two 128-bit loads per lane.
__device__ __forceinline__ v16h load_b_frag(const half_t* __restrict__ packed,
                                            int tile, int lane) {
    const half_t* p = packed + ((tile * 32 + lane) << 4);
    v16h b;
#pragma unroll
    for (int e = 0; e < 16; ++e) b[e] = p[e];
    return b;
}

#define WMMA_F16(A, B, Cacc) \
    __builtin_amdgcn_wmma_f32_16x16x32_f16(false, (A), false, (B), (short)0, (Cacc), false, false)

// ---------------------------------------------------------------------------
// prep kernels (run once per launch; tiny)
// ---------------------------------------------------------------------------
__global__ void cps_kernel(const float* __restrict__ deltas, float* __restrict__ cps) {
    int h = threadIdx.x;
    if (h < H_) {
        float tmp[C_];
        float acc = 0.0f;
#pragma unroll
        for (int c = 0; c < C_; ++c) {
            float x = deltas[h * C_ + c];
            float sp = (x > 20.0f) ? x : log1pf(__expf(x));
            acc += sp;
            tmp[c] = acc;
        }
        float sc = TIME_MAX / acc;
#pragma unroll
        for (int c = 0; c < C_; ++c) cps[h * C_ + c] = tmp[c] * sc;
    }
}

// Pack a KxN f32 row-major weight into B-fragment tile layout (f16).
// dst flat index = ((kc*(N/16)+nt)*32 + lane)*16 + e
// element source: k = kc*32 + (e>>3)*16 + (lane>>4)*8 + (e&7) ; n = nt*16 + (lane&15)
__global__ void pack_b_kernel(const float* __restrict__ src, half_t* __restrict__ dst,
                              int K, int N) {
    int i = blockIdx.x * blockDim.x + threadIdx.x;
    if (i >= K * N) return;
    int e = i & 15;
    int lane = (i >> 4) & 31;
    int t = i >> 9;
    int ntiles = N >> 4;
    int kc = t / ntiles, nt = t % ntiles;
    int k = kc * 32 + (e >> 3) * 16 + (lane >> 4) * 8 + (e & 7);
    int n = nt * 16 + (lane & 15);
    dst[i] = (half_t)src[k * N + n];
}

// Wh [H][32][32] -> per-holiday tiles (2 n-tiles each), same fragment layout
__global__ void pack_wh_kernel(const float* __restrict__ Wh, half_t* __restrict__ dst) {
    int i = blockIdx.x * blockDim.x + threadIdx.x;
    if (i >= H_ * 2 * 32 * 16) return;
    int e = i & 15;
    int lane = (i >> 4) & 31;
    int nt = (i >> 9) & 1;
    int h = i >> 10;
    int k = (e >> 3) * 16 + (lane >> 4) * 8 + (e & 7);
    int n = nt * 16 + (lane & 15);
    dst[i] = (half_t)Wh[h * 32 * 32 + k * 32 + n];
}

// ---------------------------------------------------------------------------
// fully fused mega-kernel: 32 batch rows per block, 256 threads (8 waves)
// ---------------------------------------------------------------------------
__global__ __launch_bounds__(256)
void holiday_mega(const float* __restrict__ inputs, const float* __restrict__ sku,
                  const float* __restrict__ ln_g, const float* __restrict__ ln_b,
                  const float* __restrict__ attn_w, const float* __restrict__ Wo,
                  const float* __restrict__ cps,
                  const half_t* __restrict__ whp, const half_t* __restrict__ wahp,
                  const half_t* __restrict__ waap, const half_t* __restrict__ wbp,
                  const half_t* __restrict__ wap,
                  float* __restrict__ out) {
    __shared__ __align__(16) half_t s_att[8 * ROWS_ * C_];    // 16 KB (8 holidays/pass)
    __shared__ __align__(16) half_t s_agg[ROWS_ * HID_];      // 32 KB
    __shared__ __align__(16) half_t s_h1 [ROWS_ * HID_];      // 32 KB
    __shared__ float s_pmax[8][ROWS_];                        // 1 KB
    __shared__ float s_psum[8][ROWS_];                        // 1 KB
    __shared__ float s_rmax[ROWS_], s_rinv[ROWS_], s_rout[ROWS_];

    const int tid   = threadIdx.x;
    const int lane  = tid & 31;
    const int wave  = tid >> 5;
    const int halfq = lane >> 4;
    const int nlo   = lane & 15;
    const int b0    = blockIdx.x * ROWS_;

    // ---- Phases A+B: attended (VALU) -> per-holiday WMMA+mish -> aggregated,
    //      processed 8 holidays per pass so the staging buffer stays 16 KB ----
#pragma unroll
    for (int hs = 0; hs < 2; ++hs) {
        {   // Phase A: thread = (row, holiday-in-set)
            const int row = tid & 31;
            const int h = hs * 8 + (tid >> 5);
            const float t = inputs[(b0 + row) * H_ + h];
            float f[C_];
            float mean = 0.0f;
#pragma unroll
            for (int c = 0; c < C_; ++c) {
                float v = t - cps[h * C_ + c];
                v = v > 0.0f ? v : 0.0f;
                f[c] = v;
                mean += v;
            }
            mean *= (1.0f / 32.0f);
            float var = 0.0f;
#pragma unroll
            for (int c = 0; c < C_; ++c) { float d = f[c] - mean; var += d * d; }
            var *= (1.0f / 32.0f);
            const float rstd = rsqrtf(var + LN_EPS);
#pragma unroll
            for (int c = 0; c < C_; ++c)
                f[c] = (f[c] - mean) * rstd * ln_g[h * C_ + c] + ln_b[h * C_ + c];
            const float aw = attn_w[h] * INV_TEMP;
            float mx = -3.4e38f;
#pragma unroll
            for (int c = 0; c < C_; ++c) mx = fmaxf(mx, f[c] * aw);
            float s = 0.0f;
#pragma unroll
            for (int c = 0; c < C_; ++c) s += __expf(f[c] * aw - mx);
            const float rs = fast_rcp(s);
            const int hl = tid >> 5;
#pragma unroll
            for (int c = 0; c < C_; ++c)
                s_att[hl * (ROWS_ * C_) + row * C_ + c] =
                    (half_t)(f[c] * (__expf(f[c] * aw - mx) * rs));
        }
        __syncthreads();
        {   // Phase B: wave = holiday-in-set; 2 M-tiles x 2 N-tiles
            const int h = hs * 8 + wave;
#pragma unroll
            for (int mt = 0; mt < 2; ++mt) {
                v16h a = load_a_frag_lds(s_att + wave * (ROWS_ * C_) + mt * 16 * C_,
                                         C_, 0, lane);
#pragma unroll
                for (int nt = 0; nt < 2; ++nt) {
                    v16h b = load_b_frag(whp, h * 2 + nt, lane);
                    v8f acc;
#pragma unroll
                    for (int r = 0; r < 8; ++r) acc[r] = 0.0f;
                    acc = WMMA_F16(a, b, acc);
#pragma unroll
                    for (int r = 0; r < 8; ++r) {
                        int m = mt * 16 + halfq * 8 + r;
                        s_agg[m * HID_ + h * 32 + nt * 16 + nlo] = (half_t)mish_f(acc[r]);
                    }
                }
            }
        }
        __syncthreads();
    }

    // ---- Phase C: agg_hidden = mish(aggregated @ Wah) -----------------------
    {
        v8f acc[2][4];
#pragma unroll
        for (int mt = 0; mt < 2; ++mt)
#pragma unroll
            for (int j = 0; j < 4; ++j)
#pragma unroll
                for (int r = 0; r < 8; ++r) acc[mt][j][r] = 0.0f;
#pragma unroll 4
        for (int kc = 0; kc < 16; ++kc) {
            v16h a0 = load_a_frag_lds(s_agg,             HID_, kc, lane);
            v16h a1 = load_a_frag_lds(s_agg + 16 * HID_, HID_, kc, lane);
#pragma unroll
            for (int j = 0; j < 4; ++j) {
                v16h b = load_b_frag(wahp, kc * 32 + wave * 4 + j, lane);
                acc[0][j] = WMMA_F16(a0, b, acc[0][j]);
                acc[1][j] = WMMA_F16(a1, b, acc[1][j]);
            }
        }
#pragma unroll
        for (int mt = 0; mt < 2; ++mt)
#pragma unroll
            for (int j = 0; j < 4; ++j)
#pragma unroll
                for (int r = 0; r < 8; ++r) {
                    int m = mt * 16 + halfq * 8 + r;
                    int n = (wave * 4 + j) * 16 + nlo;
                    s_h1[m * HID_ + n] = (half_t)mish_f(acc[mt][j][r]);
                }
    }
    __syncthreads();

    // ---- Phase D: logits = h1 @ Waa, kept entirely in registers -------------
    v8f accD[2][4];
#pragma unroll
    for (int mt = 0; mt < 2; ++mt)
#pragma unroll
        for (int j = 0; j < 4; ++j)
#pragma unroll
            for (int r = 0; r < 8; ++r) accD[mt][j][r] = 0.0f;
#pragma unroll 4
    for (int kc = 0; kc < 16; ++kc) {
        v16h a0 = load_a_frag_lds(s_h1,             HID_, kc, lane);
        v16h a1 = load_a_frag_lds(s_h1 + 16 * HID_, HID_, kc, lane);
#pragma unroll
        for (int j = 0; j < 4; ++j) {
            v16h b = load_b_frag(waap, kc * 32 + wave * 4 + j, lane);
            accD[0][j] = WMMA_F16(a0, b, accD[0][j]);
            accD[1][j] = WMMA_F16(a1, b, accD[1][j]);
        }
    }

    // ---- row softmax over 512 cols: lane-partial -> shuffle -> LDS partials -
#pragma unroll
    for (int mt = 0; mt < 2; ++mt)
#pragma unroll
        for (int j = 0; j < 4; ++j)
#pragma unroll
            for (int r = 0; r < 8; ++r) accD[mt][j][r] *= INV_TEMP;
    {
        float pm[2][8];
#pragma unroll
        for (int mt = 0; mt < 2; ++mt)
#pragma unroll
            for (int r = 0; r < 8; ++r) {
                float v = accD[mt][0][r];
                v = fmaxf(v, accD[mt][1][r]);
                v = fmaxf(v, accD[mt][2][r]);
                v = fmaxf(v, accD[mt][3][r]);
#pragma unroll
                for (int msk = 1; msk < 16; msk <<= 1)
                    v = fmaxf(v, __shfl_xor(v, msk, 32));
                pm[mt][r] = v;
            }
        if (nlo == 0) {
#pragma unroll
            for (int mt = 0; mt < 2; ++mt)
#pragma unroll
                for (int r = 0; r < 8; ++r)
                    s_pmax[wave][mt * 16 + halfq * 8 + r] = pm[mt][r];
        }
    }
    __syncthreads();
    if (tid < ROWS_) {
        float mx = s_pmax[0][tid];
#pragma unroll
        for (int w = 1; w < 8; ++w) mx = fmaxf(mx, s_pmax[w][tid]);
        s_rmax[tid] = mx;
        s_rout[tid] = 0.0f;
    }
    __syncthreads();
    {
        float ps[2][8];
#pragma unroll
        for (int mt = 0; mt < 2; ++mt)
#pragma unroll
            for (int r = 0; r < 8; ++r) {
                const int m = mt * 16 + halfq * 8 + r;
                const float mx = s_rmax[m];
                float acc = 0.0f;
#pragma unroll
                for (int j = 0; j < 4; ++j) {
                    float e = __expf(accD[mt][j][r] - mx);
                    accD[mt][j][r] = e;        // keep numerator in registers
                    acc += e;
                }
#pragma unroll
                for (int msk = 1; msk < 16; msk <<= 1)
                    acc += __shfl_xor(acc, msk, 32);
                ps[mt][r] = acc;
            }
        if (nlo == 0) {
#pragma unroll
            for (int mt = 0; mt < 2; ++mt)
#pragma unroll
                for (int r = 0; r < 8; ++r)
                    s_psum[wave][mt * 16 + halfq * 8 + r] = ps[mt][r];
        }
    }
    __syncthreads();
    if (tid < ROWS_) {
        float s = s_psum[0][tid];
#pragma unroll
        for (int w = 1; w < 8; ++w) s += s_psum[w][tid];
        s_rinv[tid] = fast_rcp(s);
    }
    __syncthreads();

    // ---- final: SKU projections (WMMA) + softmax-gate + output dot ----------
    {
        float racc[2][8];
#pragma unroll
        for (int mt = 0; mt < 2; ++mt)
#pragma unroll
            for (int r = 0; r < 8; ++r) racc[mt][r] = 0.0f;

        const float* skub = sku + (long)b0 * SKU_;
#pragma unroll
        for (int mt = 0; mt < 2; ++mt) {
            v16h sa0 = load_a_frag_g32(skub + mt * 16 * SKU_, SKU_, 0, lane);
            v16h sa1 = load_a_frag_g32(skub + mt * 16 * SKU_, SKU_, 1, lane);
#pragma unroll
            for (int j = 0; j < 4; ++j) {
                const int nt = wave * 4 + j;
                v8f ab, aa;
#pragma unroll
                for (int r = 0; r < 8; ++r) { ab[r] = 0.0f; aa[r] = 0.0f; }
                ab = WMMA_F16(sa0, load_b_frag(wbp, 0 * 32 + nt, lane), ab);
                ab = WMMA_F16(sa1, load_b_frag(wbp, 1 * 32 + nt, lane), ab);
                aa = WMMA_F16(sa0, load_b_frag(wap, 0 * 32 + nt, lane), aa);
                aa = WMMA_F16(sa1, load_b_frag(wap, 1 * 32 + nt, lane), aa);
                const float won = Wo[nt * 16 + nlo];
#pragma unroll
                for (int r = 0; r < 8; ++r) {
                    const int m = mt * 16 + halfq * 8 + r;
                    const float p  = accD[mt][j][r] * s_rinv[m];
                    const float hv = (float)s_h1[m * HID_ + nt * 16 + nlo];
                    racc[mt][r] += (hv * p * sigmoid_f(ab[r]) + aa[r]) * won;
                }
            }
        }
#pragma unroll
        for (int mt = 0; mt < 2; ++mt)
#pragma unroll
            for (int r = 0; r < 8; ++r) {
#pragma unroll
                for (int msk = 1; msk < 16; msk <<= 1)
                    racc[mt][r] += __shfl_xor(racc[mt][r], msk, 32);
                if (nlo == 0)
                    atomicAdd(&s_rout[mt * 16 + halfq * 8 + r], racc[mt][r]);
            }
    }
    __syncthreads();
    if (tid < ROWS_) out[b0 + tid] = s_rout[tid];
}

// ---------------------------------------------------------------------------
// launch
// ---------------------------------------------------------------------------
extern "C" void kernel_launch(void* const* d_in, const int* in_sizes, int n_in,
                              void* d_out, int out_size, void* d_ws, size_t ws_size,
                              hipStream_t stream) {
    const float* inputs = (const float*)d_in[0];
    const float* sku    = (const float*)d_in[1];
    const float* deltas = (const float*)d_in[2];
    const float* ln_g   = (const float*)d_in[3];
    const float* ln_b   = (const float*)d_in[4];
    const float* attn_w = (const float*)d_in[5];
    const float* Wh     = (const float*)d_in[6];
    const float* Wah    = (const float*)d_in[7];
    const float* Waa    = (const float*)d_in[8];
    const float* Wb     = (const float*)d_in[9];
    const float* Wa     = (const float*)d_in[10];
    const float* Wo     = (const float*)d_in[11];

    const int B = in_sizes[0] / H_;

    char* ws = (char*)d_ws;
    float*  cps  = (float*)(ws);                                   //  2 KB
    half_t* whp  = (half_t*)(ws + 8192);                           // 32 KB
    half_t* wahp = (half_t*)(ws + 65536);                          // 512 KB
    half_t* waap = (half_t*)(ws + 65536 + 524288);                 // 512 KB
    half_t* wbp  = (half_t*)(ws + 65536 + 2 * 524288);             // 64 KB
    half_t* wap  = (half_t*)(ws + 65536 + 2 * 524288 + 65536);     // 64 KB

    cps_kernel<<<1, 32, 0, stream>>>(deltas, cps);
    pack_wh_kernel<<<64, 256, 0, stream>>>(Wh, whp);
    pack_b_kernel<<<1024, 256, 0, stream>>>(Wah, wahp, 512, 512);
    pack_b_kernel<<<1024, 256, 0, stream>>>(Waa, waap, 512, 512);
    pack_b_kernel<<<128, 256, 0, stream>>>(Wb, wbp, 64, 512);
    pack_b_kernel<<<128, 256, 0, stream>>>(Wa, wap, 64, 512);

    holiday_mega<<<B / ROWS_, 256, 0, stream>>>(
        inputs, sku, ln_g, ln_b, attn_w, Wo, cps,
        whp, wahp, waap, wbp, wap, (float*)d_out);
}
